// GraphSAGE_19825569038524
// MI455X (gfx1250) — compile-verified, matrix operand order checked by reading, not run on hardware
//
#include <hip/hip_runtime.h>
#include <hip/hip_bf16.h>

// ---------------------------------------------------------------------------
// GraphSAGE (gcn aggregator), 3 layers, MI455X / gfx1250.
// project-then-aggregate + WMMA bf16 GEMM (hi/lo error-compensated split,
// pre-split into bf16 planes so the GEMM loop is pure b128-load + wmma),
// L2-resident scatter-add (192MB L2 >> P+AGG = 102MB).
// ---------------------------------------------------------------------------

typedef __attribute__((ext_vector_type(16))) __bf16 v16bf;
typedef __attribute__((ext_vector_type(8)))  __bf16 v8bf;
typedef __attribute__((ext_vector_type(4)))  __bf16 v4bf;
typedef __attribute__((ext_vector_type(8)))  float  v8f;

#define D_IN 128
#define LN_EPS 1e-5f

// ---------------------------------------------------------------------------
// Zero-fill
// ---------------------------------------------------------------------------
__global__ void zero_kernel(float* __restrict__ p, int n) {
    int i = blockIdx.x * blockDim.x + threadIdx.x;
    if (i < n) p[i] = 0.0f;
}

// ---------------------------------------------------------------------------
// deg[i] = # in-edges of node i
// ---------------------------------------------------------------------------
__global__ void degree_kernel(const int* __restrict__ dst, float* __restrict__ deg, int E) {
    int i = blockIdx.x * blockDim.x + threadIdx.x;
    int stride = gridDim.x * blockDim.x;
    for (; i < E; i += stride) atomicAdd(&deg[dst[i]], 1.0f);
}

// ---------------------------------------------------------------------------
// Split f32 -> bf16 hi/lo planes (vectorized: float4 in, 2x v4bf out)
// ---------------------------------------------------------------------------
__global__ void split_x_kernel(const float* __restrict__ X,
                               __bf16* __restrict__ H, __bf16* __restrict__ L,
                               long total) {
    long i = ((long)blockIdx.x * blockDim.x + threadIdx.x) * 4;
    if (i >= total) return;
    float4 v = *(const float4*)(X + i);
    float vv[4] = {v.x, v.y, v.z, v.w};
    v4bf h, l;
#pragma unroll
    for (int j = 0; j < 4; ++j) {
        __bf16 hj = (__bf16)vv[j];
        h[j] = hj;
        l[j] = (__bf16)(vv[j] - (float)hj);
    }
    *(v4bf*)(H + i) = h;
    *(v4bf*)(L + i) = l;
}

// ---------------------------------------------------------------------------
// Split + transpose W[k,dout] (row-major) -> WT hi/lo planes [dout,128]
// so that B fragments are K-contiguous.  Tiny (<=16K elements).
// ---------------------------------------------------------------------------
__global__ void split_w_kernel(const float* __restrict__ W,
                               __bf16* __restrict__ WH, __bf16* __restrict__ WL,
                               int dout) {
    int i = blockIdx.x * blockDim.x + threadIdx.x;
    int total = D_IN * dout;
    if (i >= total) return;
    int k  = i / dout;
    int nn = i - k * dout;
    float v = W[i];
    __bf16 h = (__bf16)v;
    WH[(long)nn * D_IN + k] = h;
    WL[(long)nn * D_IN + k] = (__bf16)(v - (float)h);
}

// ---------------------------------------------------------------------------
// P = X @ W, AGG = P (self-term seed).  One wave per 16x16 tile.
// Fragment layout (ISA 7.12.2, 16-bit A/B):
//   v16bf elements 0..7  = K: k0 + 8*half + 0..7     (one 16B chunk)
//   v16bf elements 8..15 = K: k0 + 16 + 8*half + 0..7 (one 16B chunk)
// => per K-step: 8 global_load_b128 + 3 wmma, no conversion VALU.
// ---------------------------------------------------------------------------
__global__ void wmma_gemm_kernel(const __bf16* __restrict__ XH,
                                 const __bf16* __restrict__ XL,
                                 const __bf16* __restrict__ WH,
                                 const __bf16* __restrict__ WL,
                                 float* __restrict__ P,
                                 float* __restrict__ AGG,
                                 int n, int dout, int total_tiles) {
    int wave = (blockIdx.x * blockDim.x + threadIdx.x) >> 5;
    int lane = threadIdx.x & 31;
    if (wave >= total_tiles) return;   // wave-uniform: EXEC stays all-1s for WMMA

    int ntiles_n = dout >> 4;
    int mt = wave / ntiles_n;
    int nt = wave - mt * ntiles_n;
    int m0 = mt << 4;
    int n0 = nt << 4;

    int half = lane >> 4;
    int r    = lane & 15;

    int arow = m0 + r;
    if (arow >= n) arow = n - 1;

    const __bf16* xh = XH + (long)arow * D_IN;
    const __bf16* xl = XL + (long)arow * D_IN;
    const __bf16* wh = WH + (long)(n0 + r) * D_IN;
    const __bf16* wl = WL + (long)(n0 + r) * D_IN;
    const int o0 = 8 * half;

    v8f c = {};
#pragma unroll
    for (int k0 = 0; k0 < D_IN; k0 += 32) {
        v8bf a0 = *(const v8bf*)(xh + k0 + o0);
        v8bf a1 = *(const v8bf*)(xh + k0 + 16 + o0);
        v8bf a2 = *(const v8bf*)(xl + k0 + o0);
        v8bf a3 = *(const v8bf*)(xl + k0 + 16 + o0);
        v8bf b0 = *(const v8bf*)(wh + k0 + o0);
        v8bf b1 = *(const v8bf*)(wh + k0 + 16 + o0);
        v8bf b2 = *(const v8bf*)(wl + k0 + o0);
        v8bf b3 = *(const v8bf*)(wl + k0 + 16 + o0);
        v16bf ah = __builtin_shufflevector(a0, a1, 0,1,2,3,4,5,6,7,8,9,10,11,12,13,14,15);
        v16bf al = __builtin_shufflevector(a2, a3, 0,1,2,3,4,5,6,7,8,9,10,11,12,13,14,15);
        v16bf bh = __builtin_shufflevector(b0, b1, 0,1,2,3,4,5,6,7,8,9,10,11,12,13,14,15);
        v16bf bl = __builtin_shufflevector(b2, b3, 0,1,2,3,4,5,6,7,8,9,10,11,12,13,14,15);
        // D = A*B + C with error compensation: AhBh + AhBl + AlBh (~fp32)
        c = __builtin_amdgcn_wmma_f32_16x16x32_bf16(false, ah, false, bh, (short)0, c, false, false);
        c = __builtin_amdgcn_wmma_f32_16x16x32_bf16(false, ah, false, bl, (short)0, c, false, false);
        c = __builtin_amdgcn_wmma_f32_16x16x32_bf16(false, al, false, bh, (short)0, c, false, false);
    }

    // C/D layout: VGPR rr -> M = rr + 8*half, N = lane%16
#pragma unroll
    for (int rr = 0; rr < 8; ++rr) {
        int row = m0 + rr + 8 * half;
        if (row < n) {
            long idx = (long)row * dout + (n0 + r);
            float v = c[rr];
            P[idx]   = v;
            AGG[idx] = v;
        }
    }
}

// ---------------------------------------------------------------------------
// AGG[dst] += P[src]  — one wave per edge, coalesced 128B, L2-resident.
// ---------------------------------------------------------------------------
__global__ void scatter_add_kernel(const float* __restrict__ P,
                                   const int* __restrict__ src,
                                   const int* __restrict__ dst,
                                   float* __restrict__ AGG,
                                   int E, int d) {
    int wave  = (blockIdx.x * blockDim.x + threadIdx.x) >> 5;
    int lane  = threadIdx.x & 31;
    int nwave = (gridDim.x * blockDim.x) >> 5;
    int fpl   = d >> 5;    // 4 (d=128) or 2 (d=64)

    for (int e = wave; e < E; e += nwave) {
        int s = src[e];
        int t = dst[e];
        const float* ps = P   + (long)s * d + lane * fpl;
        float*       pt = AGG + (long)t * d + lane * fpl;
        if (fpl == 4) {
            float4 v = *(const float4*)ps;
            atomicAdd(&pt[0], v.x);
            atomicAdd(&pt[1], v.y);
            atomicAdd(&pt[2], v.z);
            atomicAdd(&pt[3], v.w);
        } else {
            float2 v = *(const float2*)ps;
            atomicAdd(&pt[0], v.x);
            atomicAdd(&pt[1], v.y);
        }
    }
}

// ---------------------------------------------------------------------------
// Epilogue layers 0/1: y = AGG/(deg+1)+bias; optional EMB=y (f32);
// z = relu(LN(y)); emit z as bf16 hi/lo planes for the next layer's GEMM.
// One wave per node, 4 lanes' floats each, wave32 shuffle reduction.
// ---------------------------------------------------------------------------
__global__ void finalize_ln_kernel(const float* __restrict__ AGG,
                                   const float* __restrict__ deg,
                                   const float* __restrict__ bias,
                                   const float* __restrict__ g,
                                   const float* __restrict__ bln,
                                   __bf16* __restrict__ ZH,
                                   __bf16* __restrict__ ZL,
                                   float* __restrict__ EMB,
                                   int n) {
    int wave = (blockIdx.x * blockDim.x + threadIdx.x) >> 5;
    int lane = threadIdx.x & 31;
    if (wave >= n) return;

    float inv = 1.0f / (deg[wave] + 1.0f);
    long  base = (long)wave * D_IN + lane * 4;

    float y[4];
    float s = 0.0f, ss = 0.0f;
#pragma unroll
    for (int j = 0; j < 4; ++j) {
        float v = AGG[base + j] * inv + bias[lane * 4 + j];
        y[j] = v;
        s += v;
        ss += v * v;
    }
    if (EMB != nullptr) {
#pragma unroll
        for (int j = 0; j < 4; ++j) EMB[base + j] = y[j];
    }

#pragma unroll
    for (int m = 16; m >= 1; m >>= 1) {
        s  += __shfl_xor(s, m, 32);
        ss += __shfl_xor(ss, m, 32);
    }
    float mu   = s * (1.0f / 128.0f);
    float var  = ss * (1.0f / 128.0f) - mu * mu;
    float rstd = rsqrtf(var + LN_EPS);

    v4bf zh, zl;
#pragma unroll
    for (int j = 0; j < 4; ++j) {
        float z = (y[j] - mu) * rstd * g[lane * 4 + j] + bln[lane * 4 + j];
        z = fmaxf(z, 0.0f);
        __bf16 h = (__bf16)z;
        zh[j] = h;
        zl[j] = (__bf16)(z - (float)h);
    }
    *(v4bf*)(ZH + base) = zh;
    *(v4bf*)(ZL + base) = zl;
}

// ---------------------------------------------------------------------------
// Epilogue layer 2 (d=64): OUT = AGG/(deg+1) + bias.
// ---------------------------------------------------------------------------
__global__ void finalize_out_kernel(const float* __restrict__ AGG,
                                    const float* __restrict__ deg,
                                    const float* __restrict__ bias,
                                    float* __restrict__ OUT,
                                    int n) {
    int wave = (blockIdx.x * blockDim.x + threadIdx.x) >> 5;
    int lane = threadIdx.x & 31;
    if (wave >= n) return;

    float inv = 1.0f / (deg[wave] + 1.0f);
    long base = (long)wave * 64 + lane * 2;
#pragma unroll
    for (int j = 0; j < 2; ++j) {
        OUT[base + j] = AGG[base + j] * inv + bias[lane * 2 + j];
    }
}

// ---------------------------------------------------------------------------
// Launcher
// Inputs: 0 feats 1 src 2 dst 3 W0 4 b0 5 W1 6 b1 7 W2 8 b2
//         9 ln0_g 10 ln0_b 11 ln1_g 12 ln1_b
// Output: [ h_emb (N*128) | h (N*64) ]
// ---------------------------------------------------------------------------
extern "C" void kernel_launch(void* const* d_in, const int* in_sizes, int n_in,
                              void* d_out, int out_size, void* d_ws, size_t ws_size,
                              hipStream_t stream) {
    const float* feats = (const float*)d_in[0];
    const int*   src   = (const int*)d_in[1];
    const int*   dst   = (const int*)d_in[2];
    const float* W0    = (const float*)d_in[3];
    const float* b0    = (const float*)d_in[4];
    const float* W1    = (const float*)d_in[5];
    const float* b1    = (const float*)d_in[6];
    const float* W2    = (const float*)d_in[7];
    const float* b2    = (const float*)d_in[8];
    const float* ln0g  = (const float*)d_in[9];
    const float* ln0b  = (const float*)d_in[10];
    const float* ln1g  = (const float*)d_in[11];
    const float* ln1b  = (const float*)d_in[12];

    const int n = in_sizes[0] / D_IN;   // 100000
    const int E = in_sizes[1];          // 1600000

    float*  ws  = (float*)d_ws;
    float*  deg = ws;                                   // [n] f32
    float*  P   = deg + n;                              // [n,128] f32
    float*  AGG = P + (size_t)n * D_IN;                 // [n,128] f32
    __bf16* XH  = (__bf16*)(AGG + (size_t)n * D_IN);    // [n,128] bf16
    __bf16* XL  = XH + (size_t)n * D_IN;                // [n,128] bf16
    __bf16* WHt = XL + (size_t)n * D_IN;                // [128,128] bf16 (transposed)
    __bf16* WLt = WHt + (size_t)D_IN * D_IN;            // [128,128] bf16

    float* out_emb = (float*)d_out;                     // [n,128]
    float* out_h   = out_emb + (size_t)n * D_IN;        // [n,64]

    const int TB = 256;                  // 8 waves / block
    const int node_blocks = (n + 7) / 8; // one wave per node
    const long ntot = (long)n * D_IN;

    // degree (shared by all layers)
    zero_kernel<<<(n + TB - 1) / TB, TB, 0, stream>>>(deg, n);
    degree_kernel<<<2048, TB, 0, stream>>>(dst, deg, E);

    // split input features once
    split_x_kernel<<<(int)((ntot / 4 + TB - 1) / TB), TB, 0, stream>>>(feats, XH, XL, ntot);

    // ---- layer 0: W0 (128->128), LN0+relu -> bf16 planes
    {
        int dout = 128;
        int tiles = ((n + 15) / 16) * (dout / 16);
        split_w_kernel<<<(D_IN * dout + TB - 1) / TB, TB, 0, stream>>>(W0, WHt, WLt, dout);
        wmma_gemm_kernel<<<(tiles + 7) / 8, TB, 0, stream>>>(XH, XL, WHt, WLt, P, AGG, n, dout, tiles);
        scatter_add_kernel<<<4096, TB, 0, stream>>>(P, src, dst, AGG, E, dout);
        finalize_ln_kernel<<<node_blocks, TB, 0, stream>>>(AGG, deg, b0, ln0g, ln0b, XH, XL, nullptr, n);
    }

    // ---- layer 1: W1 (128->128), emit pre-LN emb, LN1+relu -> bf16 planes
    {
        int dout = 128;
        int tiles = ((n + 15) / 16) * (dout / 16);
        split_w_kernel<<<(D_IN * dout + TB - 1) / TB, TB, 0, stream>>>(W1, WHt, WLt, dout);
        wmma_gemm_kernel<<<(tiles + 7) / 8, TB, 0, stream>>>(XH, XL, WHt, WLt, P, AGG, n, dout, tiles);
        scatter_add_kernel<<<4096, TB, 0, stream>>>(P, src, dst, AGG, E, dout);
        finalize_ln_kernel<<<node_blocks, TB, 0, stream>>>(AGG, deg, b1, ln1g, ln1b, XH, XL, out_emb, n);
    }

    // ---- layer 2: W2 (128->64) -> out_h
    {
        int dout = 64;
        int tiles = ((n + 15) / 16) * (dout / 16);
        split_w_kernel<<<(D_IN * dout + TB - 1) / TB, TB, 0, stream>>>(W2, WHt, WLt, dout);
        wmma_gemm_kernel<<<(tiles + 7) / 8, TB, 0, stream>>>(XH, XL, WHt, WLt, P, AGG, n, dout, tiles);
        scatter_add_kernel<<<4096, TB, 0, stream>>>(P, src, dst, AGG, E, dout);
        finalize_out_kernel<<<node_blocks, TB, 0, stream>>>(AGG, deg, b2, out_h, n);
    }
}
